// TemporalRelationalLayer_4896262717869
// MI455X (gfx1250) — matrix operand order repaired and checked by reading, not compiled, not run on hardware
//
#include <hip/hip_runtime.h>
#include <math.h>
#include <stdint.h>

typedef __attribute__((ext_vector_type(16))) _Float16 v16h;
typedef __attribute__((ext_vector_type(4)))  _Float16 v4h;
typedef __attribute__((ext_vector_type(8)))  float    v8f;

#define B_  32
#define S_  1024
#define D_  512
#define H_  8
#define DK  64
#define SIGMA_ -1e10f

// A-fragment (16x32 f16) K index for element e, half hh = lane>>4
__device__ __forceinline__ int a_k(int e, int hh) {
  return e + 16 * hh;
}

// ---------------------------------------------------------------------------
// One-time f32 -> f16 weight conversion (4 elements/thread).
// ---------------------------------------------------------------------------
__global__ __launch_bounds__(256)
void cvt_f32_to_f16(const float* __restrict__ src, _Float16* __restrict__ dst, int n4)
{
  const int i = blockIdx.x * blockDim.x + threadIdx.x;
  if (i < n4) {
    const float4 f = ((const float4*)src)[i];
    v4h h;
    h[0] = (_Float16)f.x; h[1] = (_Float16)f.y;
    h[2] = (_Float16)f.z; h[3] = (_Float16)f.w;
    ((v4h*)dst)[i] = h;
  }
}

// ---------------------------------------------------------------------------
// Projection GEMM: Y[m,n] = sum_k X[m,k]*W[n,k] + bias[n]  (nn.Linear)
// X f32 [32768, 512], W f16 [512, 512] (pre-converted), out f16 qkv layout.
// One wave computes a 16(M) x 64(N) strip; grid = (256, 8).
// ---------------------------------------------------------------------------
__global__ __launch_bounds__(256)
void proj_gemm_f32(const float* __restrict__ X,
                   const _Float16* __restrict__ W,
                   const float* __restrict__ bias,
                   _Float16* __restrict__ Y)
{
  const int lane = threadIdx.x & 31;
  const int wave = threadIdx.x >> 5;
  const int hh   = lane >> 4;
  const int ln   = lane & 15;

  const int m0 = (blockIdx.x * 8 + wave) * 16;
  const int n0 = blockIdx.y * 64;

  const float* xrow = X + (size_t)(m0 + ln) * D_;

  v8f acc[4] = {};

  for (int k0 = 0; k0 < D_; k0 += 32) {
    v16h a;
    #pragma unroll
    for (int e = 0; e < 16; ++e)
      a[e] = (_Float16)xrow[k0 + a_k(e, hh)];
    #pragma unroll
    for (int t = 0; t < 4; ++t) {
      const v16h bf = *(const v16h*)&W[(size_t)(n0 + t * 16 + ln) * D_ + k0 + 16 * hh];
      acc[t] = __builtin_amdgcn_wmma_f32_16x16x32_f16(
          false, a, false, bf, (short)0, acc[t], false, false);
    }
  }

  #pragma unroll
  for (int t = 0; t < 4; ++t) {
    const int n  = n0 + t * 16 + ln;
    const float bv = bias[n];
    const int hN = n >> 6, dd = n & 63;           // n = h*64 + dd
    #pragma unroll
    for (int r = 0; r < 8; ++r) {
      const int m = m0 + r + 8 * hh;              // m = b*S + s
      const int b = m >> 10, s = m & 1023;
      Y[((size_t)((b * H_ + hN) * S_ + s)) * DK + dd] = (_Float16)(acc[t][r] + bv);
    }
  }
}

// ---------------------------------------------------------------------------
// Output GEMM: out[m,n] = sum_k A[m,k]*Wo[n,k] + bo[n]; A f16, Wo f16, out f32.
// ---------------------------------------------------------------------------
__global__ __launch_bounds__(256)
void out_gemm_f16(const _Float16* __restrict__ X,
                  const _Float16* __restrict__ W,
                  const float* __restrict__ bias,
                  float* __restrict__ Y)
{
  const int lane = threadIdx.x & 31;
  const int wave = threadIdx.x >> 5;
  const int hh   = lane >> 4;
  const int ln   = lane & 15;

  const int m0 = (blockIdx.x * 8 + wave) * 16;
  const int n0 = blockIdx.y * 64;

  const _Float16* xrow = X + (size_t)(m0 + ln) * D_;

  v8f acc[4] = {};

  for (int k0 = 0; k0 < D_; k0 += 32) {
    v16h a;
    #pragma unroll
    for (int e = 0; e < 16; ++e)
      a[e] = xrow[k0 + a_k(e, hh)];
    #pragma unroll
    for (int t = 0; t < 4; ++t) {
      const v16h bf = *(const v16h*)&W[(size_t)(n0 + t * 16 + ln) * D_ + k0 + 16 * hh];
      acc[t] = __builtin_amdgcn_wmma_f32_16x16x32_f16(
          false, a, false, bf, (short)0, acc[t], false, false);
    }
  }

  #pragma unroll
  for (int t = 0; t < 4; ++t) {
    const int n = n0 + t * 16 + ln;
    const float bv = bias[n];
    #pragma unroll
    for (int r = 0; r < 8; ++r) {
      const int m = m0 + r + 8 * hh;
      Y[(size_t)m * D_ + n] = acc[t][r] + bv;
    }
  }
}

// ---------------------------------------------------------------------------
// Flash attention with the block-diagonal exclusion mask.
// Block = 256 threads = 8 waves; wave w owns q-rows [q0, q0+16).
// K/V tiles staged memory->LDS with GLOBAL_LOAD_ASYNC_TO_LDS_B128 (ASYNCcnt).
// grid = (S/128, B*H) = (8, 256)
// ---------------------------------------------------------------------------
__global__ __launch_bounds__(256)
void attn_kernel(const _Float16* __restrict__ Q,
                 const _Float16* __restrict__ K,
                 const _Float16* __restrict__ V,
                 _Float16* __restrict__ O)
{
  __shared__ __align__(32) _Float16 Klds[64 * DK];     // [key][dim]  8KB
  __shared__ __align__(32) _Float16 Vlds[64 * DK];     // [key][dim]  8KB (row-major)
  __shared__ __align__(32) _Float16 Plds[8][16 * 64];  // per-wave P 16KB

  const int lane = threadIdx.x & 31;
  const int wave = threadIdx.x >> 5;
  const int hh   = lane >> 4;
  const int ln   = lane & 15;

  const int bh = blockIdx.y;           // b*H + h
  const int b  = bh >> 3;
  const int h  = bh & 7;
  const int q0 = blockIdx.x * 128 + wave * 16;

  const _Float16* qbase = Q + ((size_t)bh * S_ + q0) * DK;
  const _Float16* kbase = K + (size_t)bh * S_ * DK;
  const _Float16* vbase = V + (size_t)bh * S_ * DK;

  // Q fragments for dk=64 (two 16x32 A-frags), resident all kernel
  v16h qa[2];
  #pragma unroll
  for (int c = 0; c < 2; ++c)
    #pragma unroll
    for (int e = 0; e < 16; ++e)
      qa[c][e] = qbase[(size_t)ln * DK + 32 * c + a_k(e, hh)];

  v8f o[4] = {};
  float rmax[8], rsum[8];
  #pragma unroll
  for (int r = 0; r < 8; ++r) { rmax[r] = -1e30f; rsum[r] = 0.f; }

  const int ldrow = threadIdx.x >> 2;        // 0..63 key row
  const int ldseg = (threadIdx.x & 3) * 16;  // 16-half (32B) segment of dk

  const float scale = 0.125f;                // 1/sqrt(64)

  for (int kb = 0; kb < S_; kb += 64) {
    __syncthreads();
    // async K/V tile copies: global -> LDS, no VGPR staging (ASYNCcnt)
    {
      const uint32_t kl = (uint32_t)(uintptr_t)&Klds[ldrow * DK + ldseg];
      const uint64_t kg = (uint64_t)(uintptr_t)(kbase + (size_t)(kb + ldrow) * DK + ldseg);
      asm volatile("global_load_async_to_lds_b128 %0, %1, off"
                   :: "v"(kl), "v"(kg) : "memory");
      asm volatile("global_load_async_to_lds_b128 %0, %1, off offset:16"
                   :: "v"(kl), "v"(kg) : "memory");
      const uint32_t vl = (uint32_t)(uintptr_t)&Vlds[ldrow * DK + ldseg];
      const uint64_t vg = (uint64_t)(uintptr_t)(vbase + (size_t)(kb + ldrow) * DK + ldseg);
      asm volatile("global_load_async_to_lds_b128 %0, %1, off"
                   :: "v"(vl), "v"(vg) : "memory");
      asm volatile("global_load_async_to_lds_b128 %0, %1, off offset:16"
                   :: "v"(vl), "v"(vg) : "memory");
    }
    if (kb + 64 < S_) {
      __builtin_prefetch(kbase + ((size_t)(kb + 64 + ldrow)) * DK + ldseg, 0, 1);
      __builtin_prefetch(vbase + ((size_t)(kb + 64 + ldrow)) * DK + ldseg, 0, 1);
    }
    asm volatile("s_wait_asynccnt 0x0" ::: "memory");
    __syncthreads();

    // scores: 4 tiles of 16 keys, dk=64 as 2 WMMA accumulations each
    v8f sc[4];
    #pragma unroll
    for (int t = 0; t < 4; ++t) {
      v8f s = {};
      #pragma unroll
      for (int c = 0; c < 2; ++c) {
        const v16h bf = *(const v16h*)&Klds[(t * 16 + ln) * DK + 32 * c + 16 * hh];
        s = __builtin_amdgcn_wmma_f32_16x16x32_f16(
            false, qa[c], false, bf, (short)0, s, false, false);
      }
      sc[t] = s;
    }

    // mask (before scaling, as in reference) + scale
    #pragma unroll
    for (int t = 0; t < 4; ++t) {
      const int kcol = kb + t * 16 + ln;
      #pragma unroll
      for (int r = 0; r < 8; ++r) {
        const int qrow = q0 + r + 8 * hh;
        const bool same = (qrow >> 9) == (kcol >> 9);   // same 512-block
        const float msk = (same && (qrow != kcol)) ? SIGMA_ : 0.f;
        sc[t][r] = (sc[t][r] + msk) * scale;
      }
    }

    // online softmax (row reductions across 16-lane half-wave groups)
    #pragma unroll
    for (int r = 0; r < 8; ++r) {
      float tmax = fmaxf(fmaxf(sc[0][r], sc[1][r]), fmaxf(sc[2][r], sc[3][r]));
      #pragma unroll
      for (int m = 1; m < 16; m <<= 1)
        tmax = fmaxf(tmax, __shfl_xor(tmax, m, 32));
      const float newm  = fmaxf(rmax[r], tmax);
      const float alpha = __expf(rmax[r] - newm);
      rmax[r] = newm;
      float psum = 0.f;
      #pragma unroll
      for (int t = 0; t < 4; ++t) {
        const float p = __expf(sc[t][r] - newm);
        sc[t][r] = p;
        psum += p;
      }
      #pragma unroll
      for (int m = 1; m < 16; m <<= 1)
        psum += __shfl_xor(psum, m, 32);
      rsum[r] = alpha * rsum[r] + psum;
      #pragma unroll
      for (int t = 0; t < 4; ++t)
        o[t][r] *= alpha;
    }

    // C-layout -> A-layout via wave-private LDS, then P @ V
    #pragma unroll
    for (int t = 0; t < 4; ++t)
      #pragma unroll
      for (int r = 0; r < 8; ++r)
        Plds[wave][(r + 8 * hh) * 64 + t * 16 + ln] = (_Float16)sc[t][r];

    #pragma unroll
    for (int c = 0; c < 2; ++c) {
      v16h pa;
      #pragma unroll
      for (int e = 0; e < 16; ++e)
        pa[e] = Plds[wave][ln * 64 + 32 * c + a_k(e, hh)];
      #pragma unroll
      for (int t = 0; t < 4; ++t) {
        // B-frag from row-major V in LDS: element e -> V[key=32c+16hh+e][dim n]
        v16h vf;
        #pragma unroll
        for (int e = 0; e < 16; ++e)
          vf[e] = Vlds[(32 * c + 16 * hh + e) * DK + t * 16 + ln];
        o[t] = __builtin_amdgcn_wmma_f32_16x16x32_f16(
            false, pa, false, vf, (short)0, o[t], false, false);
      }
    }
  }

  // epilogue: normalize and store as [b][s][h*64+dd] f16
  #pragma unroll
  for (int t = 0; t < 4; ++t) {
    const int n = h * DK + t * 16 + ln;
    #pragma unroll
    for (int r = 0; r < 8; ++r) {
      const int qrow = q0 + r + 8 * hh;
      O[((size_t)(b * S_ + qrow)) * D_ + n] = (_Float16)(o[t][r] / rsum[r]);
    }
  }
}

// ---------------------------------------------------------------------------
extern "C" void kernel_launch(void* const* d_in, const int* in_sizes, int n_in,
                              void* d_out, int out_size, void* d_ws, size_t ws_size,
                              hipStream_t stream)
{
  const float* query = (const float*)d_in[0];
  const float* key   = (const float*)d_in[1];
  const float* value = (const float*)d_in[2];
  const float* Wq    = (const float*)d_in[3];
  const float* bq    = (const float*)d_in[4];
  const float* Wk    = (const float*)d_in[5];
  const float* bk    = (const float*)d_in[6];
  const float* Wv    = (const float*)d_in[7];
  const float* bv    = (const float*)d_in[8];
  const float* Wo    = (const float*)d_in[9];
  const float* bo    = (const float*)d_in[10];

  const size_t N  = (size_t)B_ * S_ * D_;   // 16.78M elements
  const size_t WN = (size_t)D_ * D_;        // 262144 elements
  _Float16* qws = (_Float16*)d_ws;
  _Float16* kws = qws + N;
  _Float16* vws = kws + N;
  _Float16* aws = vws + N;
  _Float16* wqh = aws + N;
  _Float16* wkh = wqh + WN;
  _Float16* wvh = wkh + WN;
  _Float16* woh = wvh + WN;

  const dim3 blk(256);
  const dim3 ggemm(256, 8);                // 2048 M-tiles / 8 waves, 8 N-strips
  const int  n4 = (int)(WN / 4);
  const dim3 gcvt((n4 + 255) / 256);

  cvt_f32_to_f16<<<gcvt, blk, 0, stream>>>(Wq, wqh, n4);
  cvt_f32_to_f16<<<gcvt, blk, 0, stream>>>(Wk, wkh, n4);
  cvt_f32_to_f16<<<gcvt, blk, 0, stream>>>(Wv, wvh, n4);
  cvt_f32_to_f16<<<gcvt, blk, 0, stream>>>(Wo, woh, n4);

  proj_gemm_f32<<<ggemm, blk, 0, stream>>>(query, wqh, bq, qws);
  proj_gemm_f32<<<ggemm, blk, 0, stream>>>(key,   wkh, bk, kws);
  proj_gemm_f32<<<ggemm, blk, 0, stream>>>(value, wvh, bv, vws);

  attn_kernel<<<dim3(8, 256), blk, 0, stream>>>(qws, kws, vws, aws);

  out_gemm_f16<<<ggemm, blk, 0, stream>>>(aws, woh, bo, (float*)d_out);
}
